// UTPM_28673201668387
// MI455X (gfx1250) — matrix operand order repaired and checked by reading, not compiled, not run on hardware
//
#include <hip/hip_runtime.h>

// ---------------- problem constants ----------------
#define E_   300
#define U_   300
#define C_   4
#define T_   8
#define N_   50
#define B_   512
#define K_   200
#define P_   44850            // E*(E-1)/2
#define PPAD 45056            // P padded to multiple of 256 (and 32)
#define EPAD 320              // E padded to multiple of 32
#define PADV (-4294967295.0f) // -2^32+1

#define NCH  (PPAD / 32)      // 1408 K-chunks of 32

typedef __attribute__((ext_vector_type(16))) __bf16          v16bf;
typedef __attribute__((ext_vector_type(16))) unsigned short  v16u;
typedef __attribute__((ext_vector_type(8)))  float           v8f;
typedef __attribute__((ext_vector_type(4)))  int             v4i;

#if __has_builtin(__builtin_amdgcn_global_load_async_to_lds_b128) && \
    __has_builtin(__builtin_amdgcn_s_wait_asynccnt)
#define USE_ASYNC_LDS 1
typedef v4i __attribute__((address_space(1)))* gv4i_p;   // global int4*
typedef v4i __attribute__((address_space(3)))* lv4i_p;   // LDS int4*
#else
#define USE_ASYNC_LDS 0
#endif

__device__ __forceinline__ unsigned short f2bf(float f) {
  unsigned int u = __builtin_bit_cast(unsigned int, f);
  unsigned int r = u + 0x7FFFu + ((u >> 16) & 1u);   // RNE
  return (unsigned short)(r >> 16);
}

// ------------- K0: pair indices + g = (v v^T)[i,j] -------------
__global__ void prep_pairs_k(const float* __restrict__ v,
                             unsigned short* __restrict__ iu,
                             unsigned short* __restrict__ ju,
                             float* __restrict__ g) {
  int j = blockIdx.x * 16 + threadIdx.x;
  int i = blockIdx.y * 16 + threadIdx.y;
  if (i >= E_ || j >= E_ || i >= j) return;
  int p = i * (E_ - 1) - (i * (i - 1)) / 2 + (j - i - 1);
  float s = 0.f;
#pragma unroll
  for (int c = 0; c < C_; ++c) s += v[i * C_ + c] * v[j * C_ + c];
  iu[p] = (unsigned short)i;
  ju[p] = (unsigned short)j;
  g[p] = s;
}

__global__ void pad_pairs_k(unsigned short* iu, unsigned short* ju, float* g) {
  int p = P_ + threadIdx.x + blockIdx.x * blockDim.x;
  if (p < PPAD) { iu[p] = 0; ju[p] = 0; g[p] = 0.f; }
}

// ------------- K1: collapse attention MLP: u = w @ w2, c = b . w2 -------------
__global__ void attn_prep_k(const float* __restrict__ w1, const float* __restrict__ b1,
                            const float* __restrict__ w2, const float* __restrict__ w3,
                            const float* __restrict__ b3, const float* __restrict__ w4,
                            float* __restrict__ u1f, float* __restrict__ u2f) {
  int e = threadIdx.x;
  if (e < E_) {
    float s1 = 0.f, s2 = 0.f;
#pragma unroll
    for (int t = 0; t < T_; ++t) {
      s1 += w1[e * T_ + t] * w2[t];
      s2 += w3[e * T_ + t] * w4[t];
    }
    u1f[e] = s1; u2f[e] = s2;
  }
  if (e == 0) {
    float c1 = 0.f, c2 = 0.f;
#pragma unroll
    for (int t = 0; t < T_; ++t) { c1 += b1[t] * w2[t]; c2 += b3[t] * w4[t]; }
    u1f[E_] = c1; u2f[E_] = c2;
  }
}

// ------------- K2: attention pooling (both heads), writes qcat[B,600] -------------
__global__ void attn_pool_k(const float* __restrict__ x,
                            const float* __restrict__ u1f, const float* __restrict__ u2f,
                            float* __restrict__ qcat) {
  __shared__ float us1[E_], us2[E_];
  __shared__ float t1[64], t2[64], p1[64], p2[64];
  int b = blockIdx.x, tid = threadIdx.x;
  for (int e = tid; e < E_; e += blockDim.x) { us1[e] = u1f[e]; us2[e] = u2f[e]; }
  __syncthreads();
  if (tid < N_) {
    const float* xr = x + ((size_t)b * N_ + tid) * E_;
    float d1 = 0.f, d2 = 0.f, as = 0.f;
    for (int e = 0; e < E_; ++e) {
      float xv = xr[e];
      d1 += xv * us1[e]; d2 += xv * us2[e]; as += fabsf(xv);
    }
    float s1 = expf(d1 + u1f[E_]);
    float s2 = expf(d2 + u2f[E_]);
    if (as == 0.f) { s1 = PADV; s2 = PADV; }
    t1[tid] = s1; t2[tid] = s2;
  }
  __syncthreads();
  if (tid == 0) {
    float m = t1[0]; for (int n = 1; n < N_; ++n) m = fmaxf(m, t1[n]);
    float s = 0.f;   for (int n = 0; n < N_; ++n) { p1[n] = expf(t1[n] - m); s += p1[n]; }
    float r = 1.f / s; for (int n = 0; n < N_; ++n) p1[n] *= r;
  }
  if (tid == 1) {
    float m = t2[0]; for (int n = 1; n < N_; ++n) m = fmaxf(m, t2[n]);
    float s = 0.f;   for (int n = 0; n < N_; ++n) { p2[n] = expf(t2[n] - m); s += p2[n]; }
    float r = 1.f / s; for (int n = 0; n < N_; ++n) p2[n] *= r;
  }
  __syncthreads();
  for (int e = tid; e < E_; e += blockDim.x) {
    float q1 = 0.f, q2 = 0.f;
    for (int n = 0; n < N_; ++n) {
      float xv = x[((size_t)b * N_ + n) * E_ + e];
      q1 += p1[n] * xv; q2 += p2[n] * xv;
    }
    qcat[(size_t)b * 600 + e]       = q1;
    qcat[(size_t)b * 600 + 300 + e] = q2;
  }
}

// ------------- K3: h = qcat@w5+b5 ; f6 = h@w6+b6  (16 batches / block) -------------
__global__ void h_f6_k(const float* __restrict__ qcat,
                       const float* __restrict__ w5, const float* __restrict__ b5,
                       const float* __restrict__ w6, const float* __restrict__ b6,
                       float* __restrict__ hbuf, float* __restrict__ f6buf) {
  __shared__ float qs[16 * 600];
  __shared__ float hs[16 * 304];
  int b0 = blockIdx.x * 16, tid = threadIdx.x;
  for (int idx = tid; idx < 16 * 600; idx += blockDim.x)
    qs[idx] = qcat[(size_t)(b0 + idx / 600) * 600 + (idx % 600)];
  __syncthreads();
  int e = tid;
  if (e < E_) {
    float acc[16];
#pragma unroll
    for (int m = 0; m < 16; ++m) acc[m] = 0.f;
    for (int k = 0; k < 600; ++k) {
      float wv = w5[k * E_ + e];
#pragma unroll
      for (int m = 0; m < 16; ++m) acc[m] += qs[m * 600 + k] * wv;
    }
    float be = b5[e];
#pragma unroll
    for (int m = 0; m < 16; ++m) {
      float hv = acc[m] + be;
      hbuf[(size_t)(b0 + m) * E_ + e] = hv;
      hs[m * 304 + e] = hv;
    }
  }
  __syncthreads();
  if (e < E_) {
    float acc[16];
#pragma unroll
    for (int m = 0; m < 16; ++m) acc[m] = 0.f;
    for (int k = 0; k < E_; ++k) {
      float wv = w6[k * E_ + e];
#pragma unroll
      for (int m = 0; m < 16; ++m) acc[m] += hs[m * 304 + k] * wv;
    }
    float be = b6[e];
#pragma unroll
    for (int m = 0; m < 16; ++m) f6buf[(size_t)(b0 + m) * E_ + e] = acc[m] + be;
  }
}

// ------------- K4: cross[b,p] = bf16(h_i * h_j * g)  -> [B, PPAD] bf16 -------------
__global__ void cross_k(const float* __restrict__ hbuf,
                        const unsigned short* __restrict__ iu,
                        const unsigned short* __restrict__ ju,
                        const float* __restrict__ g,
                        unsigned short* __restrict__ crossbf) {
  __shared__ float hs[16 * 304];
  int b0 = blockIdx.y * 16, tid = threadIdx.x;
  for (int idx = tid; idx < 16 * E_; idx += blockDim.x)
    hs[(idx / E_) * 304 + (idx % E_)] = hbuf[(size_t)(b0 + idx / E_) * E_ + (idx % E_)];
  __syncthreads();
  int p = blockIdx.x * 256 + tid;           // grid.x * 256 == PPAD
  int i = iu[p], j = ju[p];
  float gv = g[p];
#pragma unroll
  for (int m = 0; m < 16; ++m) {
    float val = hs[m * 304 + i] * hs[m * 304 + j] * gv;
    crossbf[(size_t)(b0 + m) * PPAD + p] = f2bf(val);
  }
}

// ------------- K5: w7 fp32 [P,300] -> bf16 [PPAD, EPAD] zero padded -------------
__global__ void cvt_w7_k(const float* __restrict__ w7, unsigned short* __restrict__ w7b) {
  size_t idx = (size_t)blockIdx.x * 256 + threadIdx.x;
  int k = (int)(idx / EPAD), e = (int)(idx % EPAD);
  float val = (k < P_ && e < E_) ? w7[(size_t)k * E_ + e] : 0.f;
  w7b[idx] = f2bf(val);
}

// ------------- K6: f7 = cross @ w7 + b7 via v_wmma_f32_16x16x32_bf16 -------------
// block = 128 threads (4 waves); tile BM=64 x BN=32, KC=32, double-buffered LDS.
// wave w owns M-subtile w; builds its A-frag once, does 2 WMMAs (N-subtiles 0,1).
// A-tile staged with gfx1250 async LDS loads when available (ASYNCcnt path).
__global__ void gemm_f7_k(const unsigned short* __restrict__ crossA,
                          const unsigned short* __restrict__ w7b,
                          const float* __restrict__ b7,
                          float* __restrict__ f7buf) {
  __shared__ unsigned short As[2][64][40];  // stride 40 ushorts = 80B (16B aligned rows)
  __shared__ unsigned short Bs[2][32][34];  // [n][k] transposed tile
  const int tid = threadIdx.x;
  const int lane = tid & 31, w = tid >> 5;
  const int lm = lane & 15, lh = lane >> 4;
  const int bm = blockIdx.x * 64;
  const int bn = blockIdx.y * 32;

  // staging decomposition (computed once)
  const int ar0 = tid >> 2;                 // A rows for this thread: ar0, ar0+32
  const int ao  = (tid & 3) * 8;            // 16B-aligned ushort offset within A row
  const int bk  = tid >> 2;                 // B k-row for this thread
  const int bn0 = (tid & 3) * 8;            // 8 consecutive n's

  v8f acc0 = {}, acc1 = {};
  const int aoff[8] = {0, 2, 4, 6, 16, 18, 20, 22};

  // ---- stage one K-chunk into buffer `buf` ----
  auto stage = [&](int kk, int buf) {
    // A tile: 64 x 32 bf16, 16B per transfer, 2 transfers per thread
#pragma unroll
    for (int i = 0; i < 2; ++i) {
      int row = ar0 + 32 * i;
      const unsigned short* src = crossA + (size_t)(bm + row) * PPAD + kk + ao;
      unsigned short* dst = &As[buf][row][ao];
#if USE_ASYNC_LDS
      __builtin_amdgcn_global_load_async_to_lds_b128((gv4i_p)src, (lv4i_p)dst, 0, 0);
#else
      *(uint4*)dst = *(const uint4*)src;
#endif
    }
    // B tile: 32 x 32 bf16, vector global load + transposed LDS scatter
    const unsigned short* bsrc = w7b + (size_t)(kk + bk) * EPAD + bn + bn0;
    uint4 bv = *(const uint4*)bsrc;
    const unsigned short* bu = (const unsigned short*)&bv;
#pragma unroll
    for (int c = 0; c < 8; ++c) Bs[buf][bn0 + c][bk] = bu[c];
#if !USE_ASYNC_LDS
    __builtin_prefetch(bsrc + 32 * EPAD, 0, 1);
#endif
  };

  stage(0, 0);
#if USE_ASYNC_LDS
  __builtin_amdgcn_s_wait_asynccnt(0);
#endif
  __syncthreads();

  for (int c = 0; c < NCH; ++c) {
    const int cur = c & 1;
    if (c + 1 < NCH) stage((c + 1) * 32, cur ^ 1);   // prefetch next chunk

    // A fragment (16x32 bf16): lane holds M = w*16+lm, K pairs per ISA layout
    v16u au;
    const int arow = w * 16 + lm;
#pragma unroll
    for (int q = 0; q < 8; ++q) {
      int kb = aoff[q] + lh * 8;
      au[2 * q]     = As[cur][arow][kb];
      au[2 * q + 1] = As[cur][arow][kb + 1];
    }
    v16bf afv = __builtin_bit_cast(v16bf, au);

    // two B fragments (32x16 bf16), N-subtiles 0 and 1
    {
      v16u bu;
#pragma unroll
      for (int q = 0; q < 8; ++q) {
        int kb = aoff[q] + lh * 8;
        bu[2 * q]     = Bs[cur][lm][kb];
        bu[2 * q + 1] = Bs[cur][lm][kb + 1];
      }
      v16bf bfv = __builtin_bit_cast(v16bf, bu);
      acc0 = __builtin_amdgcn_wmma_f32_16x16x32_bf16(false, afv, false, bfv,
                                                     (short)0, acc0, false, false);
    }
    {
      v16u bu;
#pragma unroll
      for (int q = 0; q < 8; ++q) {
        int kb = aoff[q] + lh * 8;
        bu[2 * q]     = Bs[cur][16 + lm][kb];
        bu[2 * q + 1] = Bs[cur][16 + lm][kb + 1];
      }
      v16bf bfv = __builtin_bit_cast(v16bf, bu);
      acc1 = __builtin_amdgcn_wmma_f32_16x16x32_bf16(false, afv, false, bfv,
                                                     (short)0, acc1, false, false);
    }

#if USE_ASYNC_LDS
    __builtin_amdgcn_s_wait_asynccnt(0);   // next chunk's A tile landed in LDS
#endif
    __syncthreads();                       // + B tile stores visible block-wide
  }

  // store D: M = bm + w*16 + r + 8*lh ; N = bn + nt*16 + lm
#pragma unroll
  for (int nt = 0; nt < 2; ++nt) {
    int col = bn + nt * 16 + lm;
    if (col < E_) {
      float bias = b7[col];
#pragma unroll
      for (int r = 0; r < 8; ++r) {
        int row = bm + w * 16 + r + 8 * lh;
        float dv = (nt == 0) ? acc0[r] : acc1[r];
        f7buf[(size_t)row * E_ + col] = dv + bias;
      }
    }
  }
}

// ------------- K7: out8 = cat(f6,f7) @ w8 + b8  (16 batches / block) -------------
__global__ void out8_k(const float* __restrict__ f6buf, const float* __restrict__ f7buf,
                       const float* __restrict__ w8, const float* __restrict__ b8,
                       float* __restrict__ o8buf) {
  __shared__ float cs[16 * 600];
  int b0 = blockIdx.x * 16, tid = threadIdx.x;
  for (int idx = tid; idx < 16 * 600; idx += blockDim.x) {
    int m = idx / 600, k = idx % 600;
    cs[idx] = (k < 300) ? f6buf[(size_t)(b0 + m) * E_ + k]
                        : f7buf[(size_t)(b0 + m) * E_ + (k - 300)];
  }
  __syncthreads();
  int e = tid;
  if (e < U_) {
    float acc[16];
#pragma unroll
    for (int m = 0; m < 16; ++m) acc[m] = 0.f;
    for (int k = 0; k < 600; ++k) {
      float wv = w8[k * U_ + e];
#pragma unroll
      for (int m = 0; m < 16; ++m) acc[m] += cs[m * 600 + k] * wv;
    }
    float be = b8[e];
#pragma unroll
    for (int m = 0; m < 16; ++m) o8buf[(size_t)(b0 + m) * U_ + e] = acc[m] + be;
  }
}

// ------------- K8: out[b,k] = out8[b,:] . label[b,k,:] -------------
__global__ void label_k(const float* __restrict__ o8buf,
                        const float* __restrict__ lab,
                        float* __restrict__ out) {
  __shared__ float os[304];
  int b = blockIdx.x, tid = threadIdx.x;
  for (int u = tid; u < U_; u += blockDim.x) os[u] = o8buf[(size_t)b * U_ + u];
  __syncthreads();
  int wave = tid >> 5, lane = tid & 31;
  for (int k = wave; k < K_; k += 8) {
    const float* lr = lab + ((size_t)b * K_ + k) * U_;
    float s = 0.f;
    for (int u = lane; u < U_; u += 32) s += os[u] * lr[u];
#pragma unroll
    for (int off = 16; off > 0; off >>= 1) s += __shfl_down(s, off, 32);
    if (lane == 0) out[(size_t)b * K_ + k] = s;
  }
}

// ============================== host launcher ==============================
extern "C" void kernel_launch(void* const* d_in, const int* in_sizes, int n_in,
                              void* d_out, int out_size, void* d_ws, size_t ws_size,
                              hipStream_t stream) {
  const float* x   = (const float*)d_in[0];
  const float* lab = (const float*)d_in[1];
  const float* w1  = (const float*)d_in[2];
  const float* b1  = (const float*)d_in[3];
  const float* w2  = (const float*)d_in[4];
  const float* w3  = (const float*)d_in[5];
  const float* b3  = (const float*)d_in[6];
  const float* w4  = (const float*)d_in[7];
  const float* w5  = (const float*)d_in[8];
  const float* b5  = (const float*)d_in[9];
  const float* w6  = (const float*)d_in[10];
  const float* b6  = (const float*)d_in[11];
  const float* w7  = (const float*)d_in[12];
  const float* b7  = (const float*)d_in[13];
  const float* w8  = (const float*)d_in[14];
  const float* b8  = (const float*)d_in[15];
  const float* v   = (const float*)d_in[16];
  float* out = (float*)d_out;

  // workspace carve-up (256B aligned)
  char* ws = (char*)d_ws;
  size_t o = 0;
  auto take = [&](size_t bytes) -> char* {
    char* p = ws + o;
    o = (o + bytes + 255) & ~(size_t)255;
    return p;
  };
  unsigned short* iu      = (unsigned short*)take((size_t)PPAD * 2);
  unsigned short* ju      = (unsigned short*)take((size_t)PPAD * 2);
  float*          g       = (float*)take((size_t)PPAD * 4);
  float*          u1f     = (float*)take(512 * 4);                 // [300]=c1
  float*          u2f     = (float*)take(512 * 4);                 // [300]=c2
  float*          qcat    = (float*)take((size_t)B_ * 600 * 4);
  float*          hbuf    = (float*)take((size_t)B_ * E_ * 4);
  float*          f6buf   = (float*)take((size_t)B_ * E_ * 4);
  float*          f7buf   = (float*)take((size_t)B_ * E_ * 4);
  float*          o8buf   = (float*)take((size_t)B_ * U_ * 4);
  unsigned short* crossbf = (unsigned short*)take((size_t)B_ * PPAD * 2);
  unsigned short* w7bf    = (unsigned short*)take((size_t)PPAD * EPAD * 2);
  (void)ws_size; (void)in_sizes; (void)n_in; (void)out_size;

  prep_pairs_k<<<dim3((E_ + 15) / 16, (E_ + 15) / 16), dim3(16, 16), 0, stream>>>(v, iu, ju, g);
  pad_pairs_k<<<1, 256, 0, stream>>>(iu, ju, g);
  attn_prep_k<<<1, 320, 0, stream>>>(w1, b1, w2, w3, b3, w4, u1f, u2f);
  attn_pool_k<<<B_, 320, 0, stream>>>(x, u1f, u2f, qcat);
  h_f6_k<<<B_ / 16, 320, 0, stream>>>(qcat, w5, b5, w6, b6, hbuf, f6buf);
  cross_k<<<dim3(PPAD / 256, B_ / 16), 256, 0, stream>>>(hbuf, iu, ju, g, crossbf);
  cvt_w7_k<<<(unsigned)(((size_t)PPAD * EPAD) / 256), 256, 0, stream>>>(w7, w7bf);
  gemm_f7_k<<<dim3(B_ / 64, EPAD / 32), 128, 0, stream>>>(crossbf, w7bf, b7, f7buf);
  out8_k<<<B_ / 16, 320, 0, stream>>>(f6buf, f7buf, w8, b8, o8buf);
  label_k<<<B_, 256, 0, stream>>>(o8buf, lab, out);
}
